// RED_GNN_trans_10763188043795
// MI455X (gfx1250) — matrix operand order repaired and checked by reading, compiled, not run on hardware
//
#include <hip/hip_runtime.h>
#include <math.h>

// ---------------------------------------------------------------------------
// RED-GNN fused message/attention/aggregation for MI455X (gfx1250, wave32).
// ---------------------------------------------------------------------------

typedef __attribute__((ext_vector_type(16))) _Float16 v16h;
typedef __attribute__((ext_vector_type(8)))  _Float16 v8h;
typedef __attribute__((ext_vector_type(8)))  float    v8f;
typedef __attribute__((ext_vector_type(4)))  float    v4f;

#define N_NODE   100000
#define N_EDGE   1280000
#define IN_DIM   64
#define OUT_DIM  64
#define ATTN_DIM 32

// Same-wave LDS ordering: DS ops are in-order per wave; the wait + memory
// clobber stops compiler reordering across the producer/consumer boundary.
#define WAVE_LDS_SYNC() asm volatile("s_wait_dscnt 0" ::: "memory")

__device__ __forceinline__ v8f wmma_f16(v16h a, v16h b, v8f c) {
  // v_wmma_f32_16x16x32_f16  (8-arg form: neg_a, A, neg_b, B, c_mod, C, reuse_a, reuse_b)
  return __builtin_amdgcn_wmma_f32_16x16x32_f16(false, a, false, b, (short)0, c,
                                                false, false);
}

// A-matrix fragment (16x32 f16, M x K) from LDS staging (row stride 64 halves).
// Lane L: m = L%16, g = L/16. halves[0..15] = K (32*ks + 16g .. +15):
// one contiguous 32-byte LDS load.
__device__ __forceinline__ v16h load_a_frag(const _Float16* xs, int m, int g, int ks) {
  return *(const v16h*)(xs + m * IN_DIM + ks * 32 + g * 16);
}

// B-matrix fragment (32x16 f16, K x N) from transposed weights W^T[n][k]
// (row stride 64 halves). Lane L: n = 16*nt + L%16, K = 32*kb + 16*(L/16) + 0..15
// -> one contiguous 32-byte LDS read.
__device__ __forceinline__ v16h load_b_frag(const _Float16* wt, int n, int g, int kb) {
  return *(const v16h*)(wt + n * IN_DIM + kb * 32 + g * 16);
}

// ---------------------------------------------------------------------------
// Kernel 0: zero the segment-sum accumulator (d_out hidden region, reused).
// ---------------------------------------------------------------------------
__global__ void __launch_bounds__(256) zero_kernel(float* __restrict__ p, int n4) {
  v4f z = {};
  for (int i = blockIdx.x * blockDim.x + threadIdx.x; i < n4;
       i += gridDim.x * blockDim.x)
    ((v4f*)p)[i] = z;
}

// ---------------------------------------------------------------------------
// Kernel 1: per-edge fused  x=(hs+hr+h_sub); msg=x@W1; attn=relu(msg@Ws);
// alpha=sigmoid(attn@w+b); atomically scatter alpha*msg into agg[obj].
// One wave per 16-edge tile, persistent grid-stride.
// ---------------------------------------------------------------------------
__global__ void __launch_bounds__(256) edge_kernel(
    const float* __restrict__ hidden, const float* __restrict__ h_sub,
    const float* __restrict__ rela,   const float* __restrict__ W1,
    const float* __restrict__ Ws,     const float* __restrict__ w_alpha_w,
    const float* __restrict__ w_alpha_b, const int* __restrict__ edges,
    float* __restrict__ agg, float* __restrict__ alpha_out)
{
  __shared__ __align__(32) _Float16 sW1T[IN_DIM * IN_DIM];     // [n][k] f16
  __shared__ __align__(32) _Float16 sWsT[ATTN_DIM * IN_DIM];   // [n][k] f16
  __shared__ __align__(32) _Float16 sX[8][16 * IN_DIM];        // per-wave staging
  __shared__ int sObj[8][16];

  const int tid = threadIdx.x;

  // Stage W1^T and Ws_attn^T as f16 once per block.
  for (int i = tid; i < IN_DIM * IN_DIM; i += 256) {
    int k = i >> 6, n = i & 63;
    sW1T[n * IN_DIM + k] = (_Float16)W1[i];
  }
  for (int i = tid; i < IN_DIM * ATTN_DIM; i += 256) {
    int k = i >> 5, n = i & 31;
    sWsT[n * IN_DIM + k] = (_Float16)Ws[i];
  }
  __syncthreads();

  const int wave = tid >> 5;
  const int lane = tid & 31;
  const int nloc = lane & 15;
  const int g    = lane >> 4;

  // Hoist weight fragments into registers (reused across all tiles).
  v16h w1f[2][4];
#pragma unroll
  for (int kb = 0; kb < 2; ++kb)
#pragma unroll
    for (int nt = 0; nt < 4; ++nt)
      w1f[kb][nt] = load_b_frag(sW1T, nt * 16 + nloc, g, kb);
  v16h wsf[2][2];
#pragma unroll
  for (int kb = 0; kb < 2; ++kb)
#pragma unroll
    for (int nt = 0; nt < 2; ++nt)
      wsf[kb][nt] = load_b_frag(sWsT, nt * 16 + nloc, g, kb);

  const float wa0  = w_alpha_w[nloc];
  const float wa1  = w_alpha_w[16 + nloc];
  const float bias = w_alpha_b[0];

  _Float16* xs = sX[wave];
  int* objs    = sObj[wave];

  const int r = lane >> 1;   // staging: 2 lanes per edge row
  const int c = lane & 1;    // half-row (32 floats)

  const int nTiles = N_EDGE / 16;
  const int waveId = blockIdx.x * 8 + wave;
  const int nWaves = gridDim.x * 8;

  for (int t = waveId; t < nTiles; t += nWaves) {
    const int ebase = t * 16;

    // ---- stage x = hidden[sub] + rela[rel] + h_sub  (f16, row stride 64) ----
    {
      const int e    = ebase + r;
      const int erel = edges[e * 6 + 2];
      const int esub = edges[e * 6 + 4];
      if (c == 0) objs[r] = edges[e * 6 + 5];
      const float* ph = hidden + (size_t)esub * IN_DIM + c * 32;
      const float* pr = rela   + (size_t)erel * IN_DIM + c * 32;
      const float* pe = h_sub  + (size_t)e    * IN_DIM + c * 32;
      _Float16* dst = xs + r * IN_DIM + c * 32;
#pragma unroll
      for (int j = 0; j < 8; ++j) {
        v4f a = *(const v4f*)(ph + j * 4);
        v4f b = *(const v4f*)(pr + j * 4);
        v4f d = *(const v4f*)(pe + j * 4);
        v4f s = a + b + d;
        dst[j * 4 + 0] = (_Float16)s[0];
        dst[j * 4 + 1] = (_Float16)s[1];
        dst[j * 4 + 2] = (_Float16)s[2];
        dst[j * 4 + 3] = (_Float16)s[3];
      }
    }
    WAVE_LDS_SYNC();

    // ---- GEMM1: msg (16x64) = X (16x64) @ W1 (64x64) ----
    v16h a0 = load_a_frag(xs, nloc, g, 0);
    v16h a1 = load_a_frag(xs, nloc, g, 1);
    v8f acc[4];
#pragma unroll
    for (int nt = 0; nt < 4; ++nt) {
      v8f z = {};
      acc[nt] = wmma_f16(a0, w1f[0][nt], z);
      acc[nt] = wmma_f16(a1, w1f[1][nt], acc[nt]);
    }

    // ---- re-stage msg as f16 A-operand for the attention GEMM ----
#pragma unroll
    for (int nt = 0; nt < 4; ++nt) {
      const int n = nt * 16 + nloc;
#pragma unroll
      for (int v = 0; v < 8; ++v) {
        const int m = v + 8 * g;
        xs[m * IN_DIM + n] = (_Float16)acc[nt][v];
      }
    }
    WAVE_LDS_SYNC();

    // ---- GEMM2: attn (16x32) = msg @ Ws_attn (64x32) ----
    v16h m0 = load_a_frag(xs, nloc, g, 0);
    v16h m1 = load_a_frag(xs, nloc, g, 1);
    v8f z0 = {}, z1 = {};
    v8f at0 = wmma_f16(m0, wsf[0][0], z0);
    at0     = wmma_f16(m1, wsf[1][0], at0);
    v8f at1 = wmma_f16(m0, wsf[0][1], z1);
    at1     = wmma_f16(m1, wsf[1][1], at1);

    // ---- alpha[m] = sigmoid(relu(attn[m]) . w_alpha + b) ----
    // attn row m lives in VGPR v=m%8 of the 16 lanes with g==m/8; reduce over
    // those 16 lanes with xor-shuffles (masks < 16 stay in-group on wave32).
    float alphas[8];
#pragma unroll
    for (int v = 0; v < 8; ++v) {
      float p = fmaxf(at0[v], 0.0f) * wa0 + fmaxf(at1[v], 0.0f) * wa1;
      p += __shfl_xor(p, 1, 32);
      p += __shfl_xor(p, 2, 32);
      p += __shfl_xor(p, 4, 32);
      p += __shfl_xor(p, 8, 32);
      alphas[v] = 1.0f / (1.0f + __expf(-(p + bias)));
    }
    if (nloc == 0) {
#pragma unroll
      for (int v = 0; v < 8; ++v)
        alpha_out[ebase + 8 * g + v] = alphas[v];
    }

    // ---- scatter: agg[obj[m]] += alpha[m] * msg[m]  (L2-resident atomics) ----
#pragma unroll
    for (int v = 0; v < 8; ++v) {
      const int m   = v + 8 * g;
      const int obj = objs[m];
      float* dst    = agg + (size_t)obj * OUT_DIM;
      const float av = alphas[v];
#pragma unroll
      for (int nt = 0; nt < 4; ++nt)
        atomicAdd(dst + nt * 16 + nloc, av * acc[nt][v]);
    }
  }
}

// ---------------------------------------------------------------------------
// Kernel 2: hidden_new = relu(agg @ W_h), computed in place on d_out rows
// (row tile depends only on its own input rows -> safe).
// ---------------------------------------------------------------------------
__global__ void __launch_bounds__(256) node_kernel(const float* __restrict__ Wh,
                                                   float* __restrict__ out)
{
  __shared__ __align__(32) _Float16 sWhT[IN_DIM * OUT_DIM];
  __shared__ __align__(32) _Float16 sX[8][16 * IN_DIM];

  const int tid = threadIdx.x;
  for (int i = tid; i < IN_DIM * OUT_DIM; i += 256) {
    int k = i >> 6, n = i & 63;
    sWhT[n * IN_DIM + k] = (_Float16)Wh[i];
  }
  __syncthreads();

  const int wave = tid >> 5;
  const int lane = tid & 31;
  const int nloc = lane & 15;
  const int g    = lane >> 4;

  v16h whf[2][4];
#pragma unroll
  for (int kb = 0; kb < 2; ++kb)
#pragma unroll
    for (int nt = 0; nt < 4; ++nt)
      whf[kb][nt] = load_b_frag(sWhT, nt * 16 + nloc, g, kb);

  _Float16* xs = sX[wave];
  const int r = lane >> 1;
  const int c = lane & 1;

  const int nTiles = N_NODE / 16;
  const int waveId = blockIdx.x * 8 + wave;
  const int nWaves = gridDim.x * 8;

  for (int t = waveId; t < nTiles; t += nWaves) {
    const int nbase = t * 16;
    {
      const float* src = out + (size_t)(nbase + r) * IN_DIM + c * 32;
      _Float16* dst = xs + r * IN_DIM + c * 32;
#pragma unroll
      for (int j = 0; j < 8; ++j) {
        v4f s = *(const v4f*)(src + j * 4);
        dst[j * 4 + 0] = (_Float16)s[0];
        dst[j * 4 + 1] = (_Float16)s[1];
        dst[j * 4 + 2] = (_Float16)s[2];
        dst[j * 4 + 3] = (_Float16)s[3];
      }
    }
    WAVE_LDS_SYNC();

    v16h a0 = load_a_frag(xs, nloc, g, 0);
    v16h a1 = load_a_frag(xs, nloc, g, 1);
#pragma unroll
    for (int nt = 0; nt < 4; ++nt) {
      v8f z = {};
      v8f acc = wmma_f16(a0, whf[0][nt], z);
      acc     = wmma_f16(a1, whf[1][nt], acc);
      const int n = nt * 16 + nloc;
#pragma unroll
      for (int v = 0; v < 8; ++v) {
        const int m = v + 8 * g;
        out[(size_t)(nbase + m) * OUT_DIM + n] = fmaxf(acc[v], 0.0f);
      }
    }
    WAVE_LDS_SYNC();  // keep next tile's staging after this tile's reads
  }
}

// ---------------------------------------------------------------------------
// Launch
// ---------------------------------------------------------------------------
extern "C" void kernel_launch(void* const* d_in, const int* in_sizes, int n_in,
                              void* d_out, int out_size, void* d_ws, size_t ws_size,
                              hipStream_t stream) {
  (void)in_sizes; (void)n_in; (void)out_size; (void)d_ws; (void)ws_size;

  const float* hidden = (const float*)d_in[0];
  const float* h_sub  = (const float*)d_in[1];
  const float* rela   = (const float*)d_in[2];
  const float* W1     = (const float*)d_in[3];
  const float* Ws     = (const float*)d_in[4];
  const float* waw    = (const float*)d_in[5];
  const float* wab    = (const float*)d_in[6];
  const float* Wh     = (const float*)d_in[7];
  const int*   edges  = (const int*)d_in[8];

  float* out       = (float*)d_out;
  float* agg       = out;                                // reuse hidden region
  float* alpha_out = out + (size_t)N_NODE * OUT_DIM;

  zero_kernel<<<1024, 256, 0, stream>>>(agg, N_NODE * OUT_DIM / 4);
  edge_kernel<<<640, 256, 0, stream>>>(hidden, h_sub, rela, W1, Ws, waw, wab,
                                       edges, agg, alpha_out);
  node_kernel<<<256, 256, 0, stream>>>(Wh, out);
}
